// MixtralDecoderLayer_76751065579772
// MI455X (gfx1250) — matrix operand order repaired and checked by reading, not compiled, not run on hardware
//
#include <hip/hip_runtime.h>
#include <math.h>

// Problem constants (from reference)
#define TT   4096
#define HH   1024
#define NHD  16
#define NKVD 4
#define DHD  64
#define IDIM 2048
#define EE   8

typedef __attribute__((ext_vector_type(16))) __bf16 v16bf;
typedef __attribute__((ext_vector_type(8)))  float  v8f;

struct __align__(16) U4 { unsigned int x, y, z, w; };

static __device__ __forceinline__ unsigned short f32_to_bf16(float f) {
    unsigned int u = __float_as_uint(f);
    u += 0x7fffu + ((u >> 16) & 1u);
    return (unsigned short)(u >> 16);
}

// Async global->LDS copy of 16 bytes (CDNA5 GLOBAL_LOAD_ASYNC_TO_LDS_B128,
// tracked by ASYNCcnt). lds_addr is the wave-relative LDS byte offset (low 32
// bits of a flat pointer to __shared__), gaddr is the 64-bit global address.
static __device__ __forceinline__ void async_copy_b128(unsigned lds_addr,
                                                       unsigned long long gaddr) {
    asm volatile("global_load_async_to_lds_b128 %0, %1, off"
                 :: "v"(lds_addr), "v"(gaddr) : "memory");
}

// ---------------------------------------------------------------------------
// WMMA bf16 GEMM:  C[M][ldc](f32) = alpha * A[M][lda](bf16) * BT[N][ldb](bf16)^T
//                                   (+ add_src if non-null)
// BT is B transposed, i.e. BT[n][k] = B[k][n], row-major with leading dim ldb.
// Block tile 256(M) x 64(N), K-chunk 32. 8 waves (256 thr) as 4(M) x 2(N),
// wave tile 64x32 = 4x2 fragments = 8 WMMAs per K-chunk.
// Double-buffered LDS with ASYNCcnt-pipelined async copies: stage n+1 is in
// flight while stage n feeds the WMMAs (wait asynccnt<=5, not 0, in steady
// state; 5 = per-wave async instructions per stage, retired in order).
// Requires: M%256==0, N%64==0, K%32==0, lda/ldb multiples of 8.
// ---------------------------------------------------------------------------
__global__ __launch_bounds__(256) void gemm_bf16_wmma(
    const unsigned short* __restrict__ A,
    const unsigned short* __restrict__ BT,
    float* __restrict__ C,
    const float* __restrict__ add_src,
    int M, int N, int K, int lda, int ldb, int ldc,
    float alpha, int causal_skip)
{
    const int blockM = blockIdx.y * 256;
    const int blockN = blockIdx.x * 64;
    // Causal skip: entire tile has s(col) > t(row) -> softmax never reads it.
    if (causal_skip && blockN > blockM + 255) return;

    __shared__ __align__(16) unsigned short shA[2][256][40]; // 32 + 8 pad
    __shared__ __align__(16) unsigned short shB[2][64][40];

    const int tid  = threadIdx.x;
    const int lane = tid & 31;
    const int wave = tid >> 5;
    const int wm   = wave & 3;   // 4 waves along M (64 rows each)
    const int wn   = wave >> 2;  // 2 waves along N (32 cols each)

    v8f acc[4][2];
    const v8f vzero = {0.f, 0.f, 0.f, 0.f, 0.f, 0.f, 0.f, 0.f};
#pragma unroll
    for (int mi = 0; mi < 4; ++mi) { acc[mi][0] = vzero; acc[mi][1] = vzero; }

    const int rlane = lane & 15;
    const int aoff  = (lane >= 16) ? 8 : 0;   // A-frag K sub-offset (elements)
    const int boff  = (lane >= 16) ? 8 : 0;   // B-frag K sub-offset (elements)

    // Uniform per-thread staging coordinates (constant trip counts, no exec
    // divergence around the WMMA region).
    const int ar = tid >> 2, aq = tid & 3;    // A: rows handled at ar + 64*ii
    const int br = tid >> 2, bq = tid & 3;    // B: one chunk per thread

    // Issue one stage of async copies (5 wave-instructions: 4 A + 1 B).
    auto stage = [&](int k0, int buf) {
#pragma unroll
        for (int ii = 0; ii < 4; ++ii) {
            int r = ar + ii * 64;
            const unsigned short* gp = A + (size_t)(blockM + r) * lda + k0 + aq * 8;
            async_copy_b128((unsigned)(uintptr_t)&shA[buf][r][aq * 8],
                            (unsigned long long)(uintptr_t)gp);
        }
        const unsigned short* gp = BT + (size_t)(blockN + br) * ldb + k0 + bq * 8;
        async_copy_b128((unsigned)(uintptr_t)&shB[buf][br][bq * 8],
                        (unsigned long long)(uintptr_t)gp);
    };

    const int nIter = K >> 5;
    stage(0, 0);

    for (int it = 0; it < nIter; ++it) {
        const int buf = it & 1;
        if (it + 1 < nIter) {
            stage((it + 1) << 5, buf ^ 1);
            // Steady state: 10 outstanding; first 5 (stage it) retire in order.
            asm volatile("s_wait_asynccnt 0x5" ::: "memory");
        } else {
            asm volatile("s_wait_asynccnt 0x0" ::: "memory");
        }
        __syncthreads();

        // A fragments: lane<16 holds row m, K {0..7,16..23}; lane>=16: K {8..15,24..31}
        v16bf afrag[4];
#pragma unroll
        for (int mi = 0; mi < 4; ++mi) {
            int row = wm * 64 + mi * 16 + rlane;
            union { U4 q[2]; v16bf v; } u;
            u.q[0] = *reinterpret_cast<const U4*>(&shA[buf][row][aoff]);
            u.q[1] = *reinterpret_cast<const U4*>(&shA[buf][row][aoff + 16]);
            afrag[mi] = u.v;
        }
        // B fragments: lane<16 holds col n, K {0..7,16..23}; lane>=16: K {8..15,24..31}
        v16bf bfrag[2];
#pragma unroll
        for (int ni = 0; ni < 2; ++ni) {
            int col = wn * 32 + ni * 16 + rlane;
            union { U4 q[2]; v16bf v; } u;
            u.q[0] = *reinterpret_cast<const U4*>(&shB[buf][col][boff]);
            u.q[1] = *reinterpret_cast<const U4*>(&shB[buf][col][boff + 16]);
            bfrag[ni] = u.v;
        }

#pragma unroll
        for (int mi = 0; mi < 4; ++mi)
#pragma unroll
            for (int ni = 0; ni < 2; ++ni)
                acc[mi][ni] = __builtin_amdgcn_wmma_f32_16x16x32_bf16(
                    false, afrag[mi], false, bfrag[ni],
                    (short)0, acc[mi][ni], false, false);

        // All waves done reading buf before anyone issues stage it+2 into it.
        __syncthreads();
    }

    // Epilogue: lane<16 -> col rlane, rows r..r+7 ; lane>=16 -> rows +8
    const int rowadd = (lane >= 16) ? 8 : 0;
#pragma unroll
    for (int mi = 0; mi < 4; ++mi) {
#pragma unroll
        for (int ni = 0; ni < 2; ++ni) {
            int col = blockN + wn * 32 + ni * 16 + rlane;
#pragma unroll
            for (int r = 0; r < 8; ++r) {
                int row = blockM + wm * 64 + mi * 16 + rowadd + r;
                float v = alpha * acc[mi][ni][r];
                if (add_src) v += add_src[(size_t)row * ldc + col];
                C[(size_t)row * ldc + col] = v;
            }
        }
    }
}

// ---------------------------------------------------------------------------
// RMSNorm: one block per row; writes bf16 (for GEMM A) and optionally f32.
// ---------------------------------------------------------------------------
__global__ __launch_bounds__(256) void rmsnorm_kernel(
    const float* __restrict__ x, const float* __restrict__ w,
    unsigned short* __restrict__ out_bf, float* __restrict__ out_f32)
{
    __shared__ float red[256];
    const int t = blockIdx.x;
    const float* row = x + (size_t)t * HH;
    float ss = 0.f;
    for (int i = threadIdx.x; i < HH; i += 256) { float v = row[i]; ss += v * v; }
    red[threadIdx.x] = ss; __syncthreads();
    for (int o = 128; o > 0; o >>= 1) {
        if (threadIdx.x < o) red[threadIdx.x] += red[threadIdx.x + o];
        __syncthreads();
    }
    float scale = rsqrtf(red[0] / (float)HH + 1e-6f);
    for (int i = threadIdx.x; i < HH; i += 256) {
        float v = row[i] * scale * w[i];
        out_bf[(size_t)t * HH + i] = f32_to_bf16(v);
        if (out_f32) out_f32[(size_t)t * HH + i] = v;
    }
}

// ---------------------------------------------------------------------------
// Transpose + f32->bf16:  in[R][Ccols] -> out[Ccols][R]
// ---------------------------------------------------------------------------
__global__ __launch_bounds__(256) void transpose_cvt_kernel(
    const float* __restrict__ in, unsigned short* __restrict__ out, int R, int Ccols)
{
    size_t idx = (size_t)blockIdx.x * 256 + threadIdx.x;
    if (idx >= (size_t)R * Ccols) return;
    int c = (int)(idx / R);
    int r = (int)(idx % R);
    out[(size_t)c * R + r] = f32_to_bf16(in[(size_t)r * Ccols + c]);
}

// ---------------------------------------------------------------------------
// RoPE + head packing. grid=(24, T), block=64.
//  head 0..15  : Q rope -> Qb[h][t][d] (bf16)
//  head 16..19 : K rope -> Kb[kv][t][d] (bf16)
//  head 20..23 : V transpose -> VTb[kv][d][t] (bf16)
// ---------------------------------------------------------------------------
__global__ __launch_bounds__(64) void rope_pack_kernel(
    const float* __restrict__ qkv,
    unsigned short* __restrict__ Qb,
    unsigned short* __restrict__ Kb,
    unsigned short* __restrict__ VTb)
{
    const int d = threadIdx.x;      // 0..63
    const int hh = blockIdx.x;      // 0..23
    const int t = blockIdx.y;       // 0..T-1
    const size_t base = (size_t)t * 1536;
    if (hh < 20) {
        const float* src = (hh < 16) ? (qkv + base + hh * DHD)
                                     : (qkv + base + 1024 + (hh - 16) * DHD);
        float x  = src[d];
        float xr = (d < 32) ? -src[d + 32] : src[d - 32];
        float invf = powf(10000.f, -((float)(2 * (d & 31))) / 64.f);
        float ang = (float)t * invf;
        float o = x * cosf(ang) + xr * sinf(ang);
        if (hh < 16)
            Qb[((size_t)hh * TT + t) * DHD + d] = f32_to_bf16(o);
        else
            Kb[((size_t)(hh - 16) * TT + t) * DHD + d] = f32_to_bf16(o);
    } else {
        int kv = hh - 20;
        float v = qkv[base + 1280 + kv * DHD + d];
        VTb[((size_t)kv * DHD + d) * TT + t] = f32_to_bf16(v);
    }
}

// ---------------------------------------------------------------------------
// Causal softmax over one row of S (already scaled by 1/sqrt(DH) via GEMM
// alpha); writes P in bf16 (zeros beyond the diagonal).
// ---------------------------------------------------------------------------
__global__ __launch_bounds__(256) void softmax_causal_kernel(
    const float* __restrict__ S, unsigned short* __restrict__ P)
{
    __shared__ float red[256];
    const int t = blockIdx.x;
    const int n = t + 1;
    const float* row = S + (size_t)t * TT;
    unsigned short* prow = P + (size_t)t * TT;

    float mx = -3.0e38f;
    for (int s = threadIdx.x; s < n; s += 256) mx = fmaxf(mx, row[s]);
    red[threadIdx.x] = mx; __syncthreads();
    for (int o = 128; o > 0; o >>= 1) {
        if (threadIdx.x < o) red[threadIdx.x] = fmaxf(red[threadIdx.x], red[threadIdx.x + o]);
        __syncthreads();
    }
    mx = red[0]; __syncthreads();

    float sum = 0.f;
    for (int s = threadIdx.x; s < n; s += 256) sum += expf(row[s] - mx);
    red[threadIdx.x] = sum; __syncthreads();
    for (int o = 128; o > 0; o >>= 1) {
        if (threadIdx.x < o) red[threadIdx.x] += red[threadIdx.x + o];
        __syncthreads();
    }
    float inv = 1.f / red[0];

    for (int s = threadIdx.x; s < TT; s += 256) {
        float p = (s < n) ? expf(row[s] - mx) * inv : 0.f;
        prow[s] = f32_to_bf16(p);
    }
}

// ---------------------------------------------------------------------------
// Elementwise f32 -> bf16
// ---------------------------------------------------------------------------
__global__ __launch_bounds__(256) void cvt_bf16_kernel(
    const float* __restrict__ in, unsigned short* __restrict__ out, size_t n)
{
    size_t i = (size_t)blockIdx.x * 256 + threadIdx.x;
    if (i < n) out[i] = f32_to_bf16(in[i]);
}

// ---------------------------------------------------------------------------
// Router: logits[t][e] = x[t]·gate_w[:,e] + gate_b[e]
// ---------------------------------------------------------------------------
__global__ __launch_bounds__(256) void router_logits_kernel(
    const float* __restrict__ x, const float* __restrict__ gw,
    const float* __restrict__ gb, float* __restrict__ logits)
{
    int idx = blockIdx.x * 256 + threadIdx.x;
    if (idx >= TT * EE) return;
    int t = idx >> 3, e = idx & 7;
    const float* xr = x + (size_t)t * HH;
    float s = gb[e];
    for (int h = 0; h < HH; ++h) s += xr[h] * gw[h * EE + e];
    logits[idx] = s;
}

// Softmax over E, top-2 (first-index tiebreak like lax.top_k), scatter weights.
__global__ __launch_bounds__(256) void router_top2_kernel(
    const float* __restrict__ logits, float* __restrict__ w_te)
{
    int t = blockIdx.x * 256 + threadIdx.x;
    if (t >= TT) return;
    float l[EE], p[EE];
    float mx = -3.0e38f;
    for (int e = 0; e < EE; ++e) { l[e] = logits[t * EE + e]; mx = fmaxf(mx, l[e]); }
    float sum = 0.f;
    for (int e = 0; e < EE; ++e) { p[e] = expf(l[e] - mx); sum += p[e]; }
    float inv = 1.f / sum;
    for (int e = 0; e < EE; ++e) p[e] *= inv;
    int e1 = 0;
    for (int e = 1; e < EE; ++e) if (p[e] > p[e1]) e1 = e;
    int e2 = (e1 == 0) ? 1 : 0;
    for (int e = 0; e < EE; ++e) if (e != e1 && p[e] > p[e2]) e2 = e;
    for (int e = 0; e < EE; ++e)
        w_te[t * EE + e] = (e == e1 || e == e2) ? p[e] : 0.f;
}

// act = bf16( silu(g1) * g3 )
__global__ __launch_bounds__(256) void silu_mul_kernel(
    const float* __restrict__ g1, const float* __restrict__ g3,
    unsigned short* __restrict__ act, size_t n)
{
    size_t i = (size_t)blockIdx.x * 256 + threadIdx.x;
    if (i >= n) return;
    float a = g1[i];
    float s = a / (1.f + expf(-a));
    act[i] = f32_to_bf16(s * g3[i]);
}

// out[t][h] += w_te[t][e] * moe_e[t][h]
__global__ __launch_bounds__(256) void moe_accum_kernel(
    float* __restrict__ out, const float* __restrict__ moe_e,
    const float* __restrict__ w_te, int e)
{
    size_t i = (size_t)blockIdx.x * 256 + threadIdx.x;
    if (i >= (size_t)TT * HH) return;
    int t = (int)(i >> 10); // HH = 1024
    out[i] += w_te[t * EE + e] * moe_e[i];
}

// ---------------------------------------------------------------------------
extern "C" void kernel_launch(void* const* d_in, const int* in_sizes, int n_in,
                              void* d_out, int out_size, void* d_ws, size_t ws_size,
                              hipStream_t stream) {
    (void)in_sizes; (void)n_in; (void)out_size; (void)ws_size;
    const float* h      = (const float*)d_in[0];
    const float* wq     = (const float*)d_in[1];
    const float* wk     = (const float*)d_in[2];
    const float* wv     = (const float*)d_in[3];
    const float* wo     = (const float*)d_in[4];
    const float* gate_w = (const float*)d_in[5];
    const float* gate_b = (const float*)d_in[6];
    const float* w1     = (const float*)d_in[7];
    const float* w2     = (const float*)d_in[8];
    const float* w3     = (const float*)d_in[9];
    const float* ln1    = (const float*)d_in[10];
    const float* ln2    = (const float*)d_in[11];
    float* out = (float*)d_out;

    char* wsb = (char*)d_ws;
    size_t off = 0;
    auto alloc = [&](size_t bytes) -> char* {
        char* p = wsb + off;
        off += (bytes + 255) & ~(size_t)255;
        return p;
    };

    unsigned short* x1b   = (unsigned short*)alloc((size_t)TT * HH * 2);
    unsigned short* qkvT  = (unsigned short*)alloc((size_t)1536 * HH * 2);
    float*          qkv   = (float*)         alloc((size_t)TT * 1536 * 4);
    unsigned short* Qb    = (unsigned short*)alloc((size_t)NHD * TT * DHD * 2);
    unsigned short* Kb    = (unsigned short*)alloc((size_t)NKVD * TT * DHD * 2);
    unsigned short* VTb   = (unsigned short*)alloc((size_t)NKVD * DHD * TT * 2);
    float*          S     = (float*)         alloc((size_t)TT * TT * 4);
    unsigned short* P     = (unsigned short*)alloc((size_t)TT * TT * 2);
    float*          attn  = (float*)         alloc((size_t)TT * HH * 4);
    unsigned short* attnb = (unsigned short*)alloc((size_t)TT * HH * 2);
    unsigned short* woT   = (unsigned short*)alloc((size_t)HH * HH * 2);
    float*          h1    = (float*)         alloc((size_t)TT * HH * 4);
    unsigned short* x2b   = (unsigned short*)alloc((size_t)TT * HH * 2);
    float*          x2f   = (float*)         alloc((size_t)TT * HH * 4);
    float*          rlog  = (float*)         alloc((size_t)TT * EE * 4);
    float*          w_te  = (float*)         alloc((size_t)TT * EE * 4);
    unsigned short* w1T   = (unsigned short*)alloc((size_t)IDIM * HH * 2);
    unsigned short* w3T   = (unsigned short*)alloc((size_t)IDIM * HH * 2);
    unsigned short* w2T   = (unsigned short*)alloc((size_t)HH * IDIM * 2);
    float*          g1    = (float*)         alloc((size_t)TT * IDIM * 4);
    float*          g3    = (float*)         alloc((size_t)TT * IDIM * 4);
    unsigned short* act   = (unsigned short*)alloc((size_t)TT * IDIM * 2);
    float*          moeE  = (float*)         alloc((size_t)TT * HH * 4);

    auto gemm = [&](const unsigned short* A, const unsigned short* BT, float* C,
                    const float* add, int M, int N, int K,
                    int lda, int ldb, int ldc, float alpha, int causal) {
        dim3 grid(N / 64, M / 256);
        gemm_bf16_wmma<<<grid, 256, 0, stream>>>(A, BT, C, add, M, N, K,
                                                 lda, ldb, ldc, alpha, causal);
    };
    auto nblk = [](size_t n) { return (unsigned)((n + 255) / 256); };

    // ---- attention branch ----
    rmsnorm_kernel<<<TT, 256, 0, stream>>>(h, ln1, x1b, nullptr);

    // Pack wq/wk/wv transposed bf16 into one [1536][1024] matrix
    transpose_cvt_kernel<<<nblk((size_t)HH * 1024), 256, 0, stream>>>(wq, qkvT, HH, 1024);
    transpose_cvt_kernel<<<nblk((size_t)HH * 256), 256, 0, stream>>>(wk, qkvT + (size_t)1024 * HH, HH, 256);
    transpose_cvt_kernel<<<nblk((size_t)HH * 256), 256, 0, stream>>>(wv, qkvT + (size_t)1280 * HH, HH, 256);

    // qkv[T][1536] = x @ [wq|wk|wv]
    gemm(x1b, qkvT, qkv, nullptr, TT, 1536, HH, HH, HH, 1536, 1.0f, 0);

    // RoPE + head packing (Q, K) and V transpose
    rope_pack_kernel<<<dim3(24, TT), 64, 0, stream>>>(qkv, Qb, Kb, VTb);

    // Per-head causal attention via GEMM + softmax + GEMM
    for (int hq = 0; hq < NHD; ++hq) {
        int kv = hq / (NHD / NKVD);
        // S = (1/sqrt(DH)) * Q_h K_h^T   (skip tiles fully above diagonal)
        gemm(Qb + (size_t)hq * TT * DHD, Kb + (size_t)kv * TT * DHD, S, nullptr,
             TT, TT, DHD, DHD, DHD, TT, 0.125f, 1);
        softmax_causal_kernel<<<TT, 256, 0, stream>>>(S, P);
        // attn[:, hq*64 : hq*64+64] = P @ V_h
        gemm(P, VTb + (size_t)kv * DHD * TT, attn + hq * DHD, nullptr,
             TT, DHD, TT, TT, TT, NHD * DHD, 1.0f, 0);
    }

    // h1 = h + attn @ wo
    cvt_bf16_kernel<<<nblk((size_t)TT * HH), 256, 0, stream>>>(attn, attnb, (size_t)TT * HH);
    transpose_cvt_kernel<<<nblk((size_t)HH * HH), 256, 0, stream>>>(wo, woT, NHD * DHD, HH);
    gemm(attnb, woT, h1, h, TT, HH, NHD * DHD, NHD * DHD, NHD * DHD, HH, 1.0f, 0);

    // ---- MoE branch ----
    rmsnorm_kernel<<<TT, 256, 0, stream>>>(h1, ln2, x2b, x2f);
    router_logits_kernel<<<nblk((size_t)TT * EE), 256, 0, stream>>>(x2f, gate_w, gate_b, rlog);
    router_top2_kernel<<<nblk((size_t)TT), 256, 0, stream>>>(rlog, w_te);

    // out starts as the residual h1
    hipMemcpyAsync(out, h1, (size_t)TT * HH * sizeof(float),
                   hipMemcpyDeviceToDevice, stream);

    for (int e = 0; e < EE; ++e) {
        const float* w1e = w1 + (size_t)e * HH * IDIM; // [H][I]
        const float* w2e = w2 + (size_t)e * IDIM * HH; // [I][H]
        const float* w3e = w3 + (size_t)e * HH * IDIM; // [H][I]
        transpose_cvt_kernel<<<nblk((size_t)HH * IDIM), 256, 0, stream>>>(w1e, w1T, HH, IDIM);
        transpose_cvt_kernel<<<nblk((size_t)HH * IDIM), 256, 0, stream>>>(w3e, w3T, HH, IDIM);
        transpose_cvt_kernel<<<nblk((size_t)IDIM * HH), 256, 0, stream>>>(w2e, w2T, IDIM, HH);

        gemm(x2b, w1T, g1, nullptr, TT, IDIM, HH, HH, HH, IDIM, 1.0f, 0);
        gemm(x2b, w3T, g3, nullptr, TT, IDIM, HH, HH, HH, IDIM, 1.0f, 0);
        silu_mul_kernel<<<nblk((size_t)TT * IDIM), 256, 0, stream>>>(g1, g3, act, (size_t)TT * IDIM);
        gemm(act, w2T, moeE, nullptr, TT, HH, IDIM, IDIM, IDIM, HH, 1.0f, 0);
        moe_accum_kernel<<<nblk((size_t)TT * HH), 256, 0, stream>>>(out, moeE, w_te, e);
    }
}
